// GPT2_26792005992740
// MI455X (gfx1250) — matrix operand order repaired and checked by reading, not compiled, not run on hardware
//
#include <hip/hip_runtime.h>
#include <hip/hip_bf16.h>
#include <math.h>

#define LYR  12
#define NH   12
#define VSZ  50257
#define SEQ  1024
#define EMB  768
#define DHD  64
#define BSZ  8
#define FF4  3072
#define QKV3 2304

typedef __bf16 bf16;
typedef __attribute__((ext_vector_type(16))) __bf16 bf16x16;
typedef __attribute__((ext_vector_type(8)))  __bf16 bf16x8;
typedef __attribute__((ext_vector_type(8)))  float  f32x8;
typedef __attribute__((ext_vector_type(4))) unsigned int u32x4;
typedef __attribute__((ext_vector_type(4))) int i32x4;
typedef __attribute__((ext_vector_type(8))) int i32x8;

#if __has_builtin(__builtin_amdgcn_tensor_load_to_lds) && \
    __has_builtin(__builtin_amdgcn_s_wait_tensorcnt)
#define GEMM_USE_TDM 1
#else
#define GEMM_USE_TDM 0
#endif

// ---------------------------------------------------------------------------
// WMMA fragment loaders (layouts per CDNA5 ISA §7.12.2).
//
// A matrix (16x32, 16-bit), row-major [M,K] source:
//   lane L holds row (L&15); element e covers K = k0 + (L>>4)*8 + e      (e<8)
//                                      K = k0 + 16 + (L>>4)*8 + (e-8)   (e>=8)
// B matrix (32x16, 16-bit), loaded from Bt row-major [N,K] source:
//   lane L holds column (L&15); element e covers K = k0 + (L>>4)*8 + e      (e<8)
//                                         K = k0 + 16 + (L>>4)*8 + (e-8)   (e>=8)
// ---------------------------------------------------------------------------
__device__ __forceinline__ bf16x16 load_fragA(const bf16* tileBase, int ld, int k0) {
  int lane = threadIdx.x & 31;
  const bf16* p = tileBase + (size_t)(lane & 15) * ld + k0 + ((lane >> 4) << 3);
  bf16x8 lo = *(const bf16x8*)p;
  bf16x8 hi = *(const bf16x8*)(p + 16);
  return __builtin_shufflevector(lo, hi, 0,1,2,3,4,5,6,7,8,9,10,11,12,13,14,15);
}

__device__ __forceinline__ bf16x16 load_fragB(const bf16* tileBase, int ld, int k0) {
  int lane = threadIdx.x & 31;
  const bf16* p = tileBase + (size_t)(lane & 15) * ld + k0 + ((lane >> 4) << 3);
  bf16x8 lo = *(const bf16x8*)p;
  bf16x8 hi = *(const bf16x8*)(p + 16);
  return __builtin_shufflevector(lo, hi, 0,1,2,3,4,5,6,7,8,9,10,11,12,13,14,15);
}

__device__ __forceinline__ f32x8 wmma_bf16(bf16x16 a, bf16x16 b, f32x8 c) {
  return __builtin_amdgcn_wmma_f32_16x16x32_bf16(false, a, false, b, (short)0, c,
                                                 false, false);
}

__device__ __forceinline__ float gelu_f(float x) {
  float x3 = x * x * x;
  return 0.5f * x * (1.0f + tanhf(0.7978845608028654f * (x + 0.044715f * x3)));
}

#if GEMM_USE_TDM
// ---------------------------------------------------------------------------
// TDM: DMA a 128-row x 32-col bf16 tile (row stride ldElems) from global into
// LDS at byte offset ldsOff, packed [128][32].  D# per CDNA5 ISA.
// ---------------------------------------------------------------------------
__device__ __forceinline__ void tdm_load_tile_128x32(unsigned ldsOff,
                                                     const bf16* gsrc, int ldElems) {
  unsigned long long ga = (unsigned long long)(const void*)gsrc;
  u32x4 g0;
  g0[0] = 1u;                                               // count=1 (valid D#)
  g0[1] = ldsOff;                                           // lds_addr (bytes)
  g0[2] = (unsigned)ga;                                     // global_addr[31:0]
  g0[3] = (unsigned)((ga >> 32) & 0x1ffffffu) | (2u << 30); // addr[56:32] | type=2
  unsigned td0 = (unsigned)ldElems;                         // tensor_dim0 (elems)
  unsigned td1 = 128u;                                      // tensor_dim1 (rows)
  i32x8 g1;
  g1[0] = (int)(1u << 16);                                  // data_size = 2 bytes
  g1[1] = (int)((td0 & 0xffffu) << 16);                     // tensor_dim0 lo16
  g1[2] = (int)(((td0 >> 16) & 0xffffu) | ((td1 & 0xffffu) << 16));
  g1[3] = (int)(((td1 >> 16) & 0xffffu) | (32u << 16));     // tile_dim0 = 32
  g1[4] = (int)128;                                         // tile_dim1 = 128
  g1[5] = (int)(unsigned)ldElems;                           // dim0_stride lo32
  g1[6] = 0;                                                // stride hi16 | dim1_stride lo
  g1[7] = 0;
  i32x4 z4 = (i32x4){0, 0, 0, 0};
#if __clang_major__ >= 23
  i32x8 z8 = (i32x8){0, 0, 0, 0, 0, 0, 0, 0};
  __builtin_amdgcn_tensor_load_to_lds(g0, g1, z4, z4, z8, 0);
#else
  __builtin_amdgcn_tensor_load_to_lds(g0, g1, z4, z4, 0);
#endif
}
#endif

// ---------------------------------------------------------------------------
// Weight convert + transpose: src fp32 [K,N] row-major -> dst bf16 [N,K].
// ---------------------------------------------------------------------------
__global__ void convT_kernel(const float* __restrict__ src, bf16* __restrict__ dst,
                             int K, int N) {
  size_t per = (size_t)K * N;
  const float* s = src + per * blockIdx.y;
  bf16* d = dst + per * blockIdx.y;
  for (size_t i = (size_t)blockIdx.x * blockDim.x + threadIdx.x; i < per;
       i += (size_t)gridDim.x * blockDim.x) {
    int k = (int)(i / N), n = (int)(i % N);
    d[(size_t)n * K + k] = (bf16)s[i];
  }
}

// ---------------------------------------------------------------------------
// Embedding: h[row] = tok_emb[x[row]] + pos_emb[s]
// ---------------------------------------------------------------------------
__global__ void embed_kernel(const int* __restrict__ x, const float* __restrict__ tok,
                             const float* __restrict__ pos, float* __restrict__ h) {
  int row = blockIdx.x;                 // 0..8191
  int s = row & (SEQ - 1);
  int t = x[row];
  const float* tr = tok + (size_t)t * EMB;
  const float* pr = pos + (size_t)s * EMB;
  float* hr = h + (size_t)row * EMB;
  for (int c = threadIdx.x; c < EMB; c += blockDim.x) hr[c] = tr[c] + pr[c];
}

// ---------------------------------------------------------------------------
// LayerNorm over E=768, one block (256 thr) per row, bf16 output.
// ---------------------------------------------------------------------------
__global__ void ln_bf16_kernel(const float* __restrict__ x, const float* __restrict__ g,
                               const float* __restrict__ bta, bf16* __restrict__ out) {
  int row = blockIdx.x;
  int tid = threadIdx.x;
  const float* xr = x + (size_t)row * EMB;
  __shared__ float red[256];
  float v0 = xr[tid], v1 = xr[tid + 256], v2 = xr[tid + 512];
  red[tid] = v0 + v1 + v2;
  __syncthreads();
  for (int o = 128; o > 0; o >>= 1) { if (tid < o) red[tid] += red[tid + o]; __syncthreads(); }
  float mu = red[0] * (1.0f / EMB);
  __syncthreads();
  float d0 = v0 - mu, d1 = v1 - mu, d2 = v2 - mu;
  red[tid] = d0 * d0 + d1 * d1 + d2 * d2;
  __syncthreads();
  for (int o = 128; o > 0; o >>= 1) { if (tid < o) red[tid] += red[tid + o]; __syncthreads(); }
  float rstd = rsqrtf(red[0] * (1.0f / EMB) + 1e-5f);
  bf16* orow = out + (size_t)row * EMB;
  orow[tid]       = (bf16)(g[tid]       * d0 * rstd + bta[tid]);
  orow[tid + 256] = (bf16)(g[tid + 256] * d1 * rstd + bta[tid + 256]);
  orow[tid + 512] = (bf16)(g[tid + 512] * d2 * rstd + bta[tid + 512]);
}

// Final LN for the 8 last-token rows only, fp32 output.
__global__ void ln_last_kernel(const float* __restrict__ x, const float* __restrict__ g,
                               const float* __restrict__ bta, float* __restrict__ out) {
  int b = blockIdx.x;                       // 0..7
  int row = b * SEQ + (SEQ - 1);
  int tid = threadIdx.x;
  const float* xr = x + (size_t)row * EMB;
  __shared__ float red[256];
  float v0 = xr[tid], v1 = xr[tid + 256], v2 = xr[tid + 512];
  red[tid] = v0 + v1 + v2;
  __syncthreads();
  for (int o = 128; o > 0; o >>= 1) { if (tid < o) red[tid] += red[tid + o]; __syncthreads(); }
  float mu = red[0] * (1.0f / EMB);
  __syncthreads();
  float d0 = v0 - mu, d1 = v1 - mu, d2 = v2 - mu;
  red[tid] = d0 * d0 + d1 * d1 + d2 * d2;
  __syncthreads();
  for (int o = 128; o > 0; o >>= 1) { if (tid < o) red[tid] += red[tid + o]; __syncthreads(); }
  float rstd = rsqrtf(red[0] * (1.0f / EMB) + 1e-5f);
  float* orow = out + (size_t)b * EMB;
  orow[tid]       = g[tid]       * d0 * rstd + bta[tid];
  orow[tid + 256] = g[tid + 256] * d1 * rstd + bta[tid + 256];
  orow[tid + 512] = g[tid + 512] * d2 * rstd + bta[tid + 512];
}

// ---------------------------------------------------------------------------
// WMMA GEMM:  C[M,N] = A[M,K](bf16) * Bt[N,K]^T(bf16) + bias
//   MODE 0: Cf = result                   (fp32 out)
//   MODE 1: Cf += result                  (residual accumulate, fp32)
//   MODE 2: Cb = bf16(gelu(result))       (MLP intermediate)
// Block = 256 thr = 8 waves (4 along M x 2 along N); block tile 128x128;
// wave tile 32x64 (2x4 WMMA accumulators).  M%128==0, N%128==0, K%32==0.
// A/B K-slices are staged into LDS (TDM double-buffered when available) so
// the 8 waves share one copy of each 128x32 tile instead of re-fetching.
// ---------------------------------------------------------------------------
template <int MODE>
__global__ void gemm_wmma_kernel(const bf16* __restrict__ A, const bf16* __restrict__ Bt,
                                 const float* __restrict__ bias, float* __restrict__ Cf,
                                 bf16* __restrict__ Cb, int M, int N, int K) {
  __shared__ __align__(16) bf16 lA[2][128 * 32];
  __shared__ __align__(16) bf16 lB[2][128 * 32];
  int lane = threadIdx.x & 31;
  int wave = threadIdx.x >> 5;
  int wm = wave & 3, wn = wave >> 2;
  int nn = lane & 15, grp = lane >> 4;
  int blockRow = blockIdx.y * 128;
  int blockCol = blockIdx.x * 128;
  int rowBase = blockRow + wm * 32;
  int colBase = blockCol + wn * 64;
  const bf16* Ablk = A + (size_t)blockRow * K;
  const bf16* Bblk = Bt + (size_t)blockCol * K;

  f32x8 acc[2][4];
#pragma unroll
  for (int j = 0; j < 4; ++j) {
    float bv = bias[colBase + j * 16 + nn];
#pragma unroll
    for (int i = 0; i < 2; ++i) {
#pragma unroll
      for (int e = 0; e < 8; ++e) acc[i][j][e] = bv;
    }
  }

  int nsteps = K / 32;

#if GEMM_USE_TDM
  unsigned ldsA0 = (unsigned)(unsigned long long)(void*)&lA[0][0];
  unsigned ldsA1 = (unsigned)(unsigned long long)(void*)&lA[1][0];
  unsigned ldsB0 = (unsigned)(unsigned long long)(void*)&lB[0][0];
  unsigned ldsB1 = (unsigned)(unsigned long long)(void*)&lB[1][0];
  if (wave == 0) {                       // one wave drives the TDM for the block
    tdm_load_tile_128x32(ldsA0, Ablk, K);
    tdm_load_tile_128x32(ldsB0, Bblk, K);
  }
  for (int i = 0; i < nsteps; ++i) {
    int buf = i & 1;
    __syncthreads();                     // prior reads of buf^1 complete
    if (wave == 0) {
      if (i + 1 < nsteps) {              // prefetch next slice via TDM
        tdm_load_tile_128x32(buf ? ldsA0 : ldsA1, Ablk + (i + 1) * 32, K);
        tdm_load_tile_128x32(buf ? ldsB0 : ldsB1, Bblk + (i + 1) * 32, K);
        __builtin_amdgcn_s_wait_tensorcnt(2);   // oldest pair (buf) has landed
      } else {
        __builtin_amdgcn_s_wait_tensorcnt(0);
      }
    }
    __syncthreads();                     // buf data visible to all waves
    bf16x16 af0 = load_fragA(&lA[buf][(wm * 32) * 32], 32, 0);
    bf16x16 af1 = load_fragA(&lA[buf][(wm * 32 + 16) * 32], 32, 0);
#pragma unroll
    for (int j = 0; j < 4; ++j) {
      bf16x16 bfj = load_fragB(&lB[buf][(wn * 64 + j * 16) * 32], 32, 0);
      acc[0][j] = wmma_bf16(af0, bfj, acc[0][j]);
      acc[1][j] = wmma_bf16(af1, bfj, acc[1][j]);
    }
  }
#else
  for (int i = 0; i < nsteps; ++i) {
    int k0 = i * 32;
    __syncthreads();
    // cooperative copy: 128 rows x 32 cols = 512 8-elem chunks, 2 per thread
    for (int c = threadIdx.x; c < 512; c += 256) {
      int r = c >> 2, w = (c & 3) * 8;
      *(bf16x8*)&lA[0][r * 32 + w] = *(const bf16x8*)&Ablk[(size_t)r * K + k0 + w];
      *(bf16x8*)&lB[0][r * 32 + w] = *(const bf16x8*)&Bblk[(size_t)r * K + k0 + w];
    }
    __syncthreads();
    bf16x16 af0 = load_fragA(&lA[0][(wm * 32) * 32], 32, 0);
    bf16x16 af1 = load_fragA(&lA[0][(wm * 32 + 16) * 32], 32, 0);
#pragma unroll
    for (int j = 0; j < 4; ++j) {
      bf16x16 bfj = load_fragB(&lB[0][(wn * 64 + j * 16) * 32], 32, 0);
      acc[0][j] = wmma_bf16(af0, bfj, acc[0][j]);
      acc[1][j] = wmma_bf16(af1, bfj, acc[1][j]);
    }
  }
#endif

#pragma unroll
  for (int i = 0; i < 2; ++i) {
#pragma unroll
    for (int j = 0; j < 4; ++j) {
#pragma unroll
      for (int v = 0; v < 8; ++v) {
        int row = rowBase + i * 16 + v + 8 * grp;
        int col = colBase + j * 16 + nn;
        size_t idx = (size_t)row * N + col;
        float val = acc[i][j][v];
        if (MODE == 0)      Cf[idx] = val;
        else if (MODE == 1) Cf[idx] = Cf[idx] + val;
        else                Cb[idx] = (bf16)gelu_f(val);
      }
    }
  }
}

// ---------------------------------------------------------------------------
// QKV split / layout prep:
//   qkv fp32 [B*S, 2304] ->  Q bf16 [B,H,S,64] (scaled by 1/sqrt(64))
//                            K bf16 [B,H,S,64]
//                            Vt bf16 [B,H,64,S]   (transposed for P*V WMMA)
// ---------------------------------------------------------------------------
__global__ void attn_prep_kernel(const float* __restrict__ qkv, bf16* __restrict__ Q,
                                 bf16* __restrict__ Kc, bf16* __restrict__ Vt) {
  int row = blockIdx.x;                  // 0..8191
  int b = row >> 10, s = row & (SEQ - 1);
  const float* qr = qkv + (size_t)row * QKV3;
  for (int c = threadIdx.x; c < QKV3; c += blockDim.x) {
    float val = qr[c];
    int part = c / EMB, e = c % EMB;
    int hh = e >> 6, d = e & 63;
    size_t hb = (size_t)(b * NH + hh);
    if (part == 0)      Q[(hb * SEQ + s) * DHD + d] = (bf16)(val * 0.125f);
    else if (part == 1) Kc[(hb * SEQ + s) * DHD + d] = (bf16)val;
    else                Vt[(hb * DHD + d) * SEQ + s] = (bf16)val;
  }
}

// ---------------------------------------------------------------------------
// Flash attention (causal), one wave per (b, h, 16-row q tile).
// Scores via WMMA; online softmax with 16-lane shuffle reductions; private
// LDS bounce converts the probability tile from C layout to the A-fragment
// layout for the P*V WMMA against Vt.  Score matrices never touch HBM.
// ---------------------------------------------------------------------------
__global__ void attention_kernel(const bf16* __restrict__ Q, const bf16* __restrict__ Kc,
                                 const bf16* __restrict__ Vt, bf16* __restrict__ ctxOut) {
  int lane = threadIdx.x & 31;
  int qt = blockIdx.x, hh = blockIdx.y, b = blockIdx.z;
  int nn = lane & 15, grp = lane >> 4;
  __shared__ __align__(16) bf16 pls[16 * 32];

  const bf16* Qb = Q + ((size_t)(b * NH + hh) * SEQ + (size_t)qt * 16) * DHD;
  const bf16* Kb = Kc + (size_t)(b * NH + hh) * SEQ * DHD;
  const bf16* Vb = Vt + (size_t)(b * NH + hh) * DHD * SEQ;

  bf16x16 qa0 = load_fragA(Qb, DHD, 0);
  bf16x16 qa1 = load_fragA(Qb, DHD, 32);

  f32x8 o[4];
  float mrow[8], lrow[8];
#pragma unroll
  for (int t = 0; t < 4; ++t)
#pragma unroll
    for (int e = 0; e < 8; ++e) o[t][e] = 0.0f;
#pragma unroll
  for (int v = 0; v < 8; ++v) { mrow[v] = -3.0e38f; lrow[v] = 0.0f; }

  int nkb = (qt >> 1) + 1;               // 32-key blocks needed (causal)
  for (int kb = 0; kb < nkb; ++kb) {
    int k0 = kb * 32;
    f32x8 st[2];
#pragma unroll
    for (int hf = 0; hf < 2; ++hf) {
      int n0 = k0 + hf * 16;
      f32x8 s;
#pragma unroll
      for (int e = 0; e < 8; ++e) s[e] = 0.0f;
      bf16x16 kf0 = load_fragB(Kb + (size_t)n0 * DHD, DHD, 0);
      bf16x16 kf1 = load_fragB(Kb + (size_t)n0 * DHD, DHD, 32);
      s = wmma_bf16(qa0, kf0, s);
      s = wmma_bf16(qa1, kf1, s);
#pragma unroll
      for (int v = 0; v < 8; ++v) {       // causal mask
        int q = qt * 16 + v + 8 * grp;
        int key = n0 + nn;
        if (key > q) s[v] = -3.0e38f;
      }
      st[hf] = s;
    }
    float scale[8];
#pragma unroll
    for (int v = 0; v < 8; ++v) {
      float x = fmaxf(st[0][v], st[1][v]);
      x = fmaxf(x, __shfl_xor(x, 8, 16));
      x = fmaxf(x, __shfl_xor(x, 4, 16));
      x = fmaxf(x, __shfl_xor(x, 2, 16));
      x = fmaxf(x, __shfl_xor(x, 1, 16));
      float nm = fmaxf(mrow[v], x);
      scale[v] = __expf(mrow[v] - nm);
      mrow[v] = nm;
      st[0][v] = __expf(st[0][v] - nm);
      st[1][v] = __expf(st[1][v] - nm);
      float r = st[0][v] + st[1][v];
      r += __shfl_xor(r, 8, 16);
      r += __shfl_xor(r, 4, 16);
      r += __shfl_xor(r, 2, 16);
      r += __shfl_xor(r, 1, 16);
      lrow[v] = lrow[v] * scale[v] + r;
    }
#pragma unroll
    for (int t = 0; t < 4; ++t)
#pragma unroll
      for (int v = 0; v < 8; ++v) o[t][v] *= scale[v];

    // C-layout -> A-fragment layout via wave-private LDS (in-order per wave)
#pragma unroll
    for (int hf = 0; hf < 2; ++hf)
#pragma unroll
      for (int v = 0; v < 8; ++v)
        pls[(v + 8 * grp) * 32 + hf * 16 + nn] = (bf16)st[hf][v];

    bf16x16 pf = load_fragA(pls, 32, 0);
#pragma unroll
    for (int t = 0; t < 4; ++t) {
      bf16x16 vf = load_fragB(Vb + (size_t)(t * 16) * SEQ, SEQ, k0);
      o[t] = wmma_bf16(pf, vf, o[t]);
    }
  }

#pragma unroll
  for (int v = 0; v < 8; ++v) lrow[v] = 1.0f / lrow[v];
#pragma unroll
  for (int t = 0; t < 4; ++t) {
#pragma unroll
    for (int v = 0; v < 8; ++v) {
      int rowg = b * SEQ + qt * 16 + v + 8 * grp;
      int col = hh * DHD + t * 16 + nn;
      ctxOut[(size_t)rowg * EMB + col] = (bf16)(o[t][v] * lrow[v]);
    }
  }
}

// ---------------------------------------------------------------------------
// Logits: out[b][v] = dot(last[b], tok_emb[v]).  Memory bound: stream tok_emb
// once (154 MB -> ~6.6 us at 23.3 TB/s); the 8x768 activations live in LDS.
// ---------------------------------------------------------------------------
__global__ void logits_kernel(const float* __restrict__ last, const float* __restrict__ tok,
                              float* __restrict__ out) {
  __shared__ float sl[BSZ * EMB];
  for (int i = threadIdx.x; i < BSZ * EMB; i += blockDim.x) sl[i] = last[i];
  __syncthreads();
  int v = blockIdx.x * blockDim.x + threadIdx.x;
  if (v >= VSZ) return;
  const float* tr = tok + (size_t)v * EMB;
  float acc[BSZ];
#pragma unroll
  for (int b = 0; b < BSZ; ++b) acc[b] = 0.0f;
  for (int e = 0; e < EMB; ++e) {
    float w = tr[e];
#pragma unroll
    for (int b = 0; b < BSZ; ++b) acc[b] += sl[b * EMB + e] * w;
  }
#pragma unroll
  for (int b = 0; b < BSZ; ++b) out[(size_t)b * VSZ + v] = acc[b];
}

// ---------------------------------------------------------------------------
// Host orchestration
// ---------------------------------------------------------------------------
extern "C" void kernel_launch(void* const* d_in, const int* in_sizes, int n_in,
                              void* d_out, int out_size, void* d_ws, size_t ws_size,
                              hipStream_t stream) {
  const int*   x      = (const int*)d_in[0];
  const float* tok    = (const float*)d_in[1];
  const float* pos    = (const float*)d_in[2];
  const float* ln1_g  = (const float*)d_in[3];
  const float* ln1_b  = (const float*)d_in[4];
  const float* Wqkv   = (const float*)d_in[5];
  const float* bqkv   = (const float*)d_in[6];
  const float* Wo     = (const float*)d_in[7];
  const float* bo     = (const float*)d_in[8];
  const float* ln2_g  = (const float*)d_in[9];
  const float* ln2_b  = (const float*)d_in[10];
  const float* W1     = (const float*)d_in[11];
  const float* b1     = (const float*)d_in[12];
  const float* W2     = (const float*)d_in[13];
  const float* b2     = (const float*)d_in[14];
  const float* lnf_g  = (const float*)d_in[15];
  const float* lnf_b  = (const float*)d_in[16];
  (void)in_sizes; (void)n_in; (void)out_size; (void)ws_size;

  size_t off = 0;
  auto alloc = [&](size_t bytes) -> void* {
    void* p = (char*)d_ws + off;
    off += (bytes + 255) & ~(size_t)255;
    return p;
  };
  bf16*  wqkvT = (bf16*)alloc((size_t)LYR * QKV3 * EMB * sizeof(bf16));
  bf16*  woT   = (bf16*)alloc((size_t)LYR * EMB * EMB * sizeof(bf16));
  bf16*  w1T   = (bf16*)alloc((size_t)LYR * FF4 * EMB * sizeof(bf16));
  bf16*  w2T   = (bf16*)alloc((size_t)LYR * EMB * FF4 * sizeof(bf16));
  float* hbuf  = (float*)alloc((size_t)BSZ * SEQ * EMB * sizeof(float));
  bf16*  aB    = (bf16*)alloc((size_t)BSZ * SEQ * EMB * sizeof(bf16));
  float* qkvB  = (float*)alloc((size_t)BSZ * SEQ * QKV3 * sizeof(float));
  bf16*  Qb    = (bf16*)alloc((size_t)BSZ * NH * SEQ * DHD * sizeof(bf16));
  bf16*  Kb    = (bf16*)alloc((size_t)BSZ * NH * SEQ * DHD * sizeof(bf16));
  bf16*  Vtb   = (bf16*)alloc((size_t)BSZ * NH * DHD * SEQ * sizeof(bf16));
  bf16*  ffB   = (bf16*)alloc((size_t)BSZ * SEQ * FF4 * sizeof(bf16));
  float* lastB = (float*)alloc((size_t)BSZ * EMB * sizeof(float));

  const int ROWS = BSZ * SEQ;  // 8192

  // Weight convert + transpose (bf16, [N,K] row-major)
  convT_kernel<<<dim3(6912, LYR), 256, 0, stream>>>(Wqkv, wqkvT, EMB, QKV3);
  convT_kernel<<<dim3(2304, LYR), 256, 0, stream>>>(Wo,   woT,   EMB, EMB);
  convT_kernel<<<dim3(9216, LYR), 256, 0, stream>>>(W1,   w1T,   EMB, FF4);
  convT_kernel<<<dim3(9216, LYR), 256, 0, stream>>>(W2,   w2T,   FF4, EMB);

  embed_kernel<<<ROWS, 256, 0, stream>>>(x, tok, pos, hbuf);

  for (int l = 0; l < LYR; ++l) {
    ln_bf16_kernel<<<ROWS, 256, 0, stream>>>(hbuf, ln1_g + (size_t)l * EMB,
                                             ln1_b + (size_t)l * EMB, aB);
    gemm_wmma_kernel<0><<<dim3(QKV3 / 128, ROWS / 128), 256, 0, stream>>>(
        aB, wqkvT + (size_t)l * QKV3 * EMB, bqkv + (size_t)l * QKV3,
        qkvB, nullptr, ROWS, QKV3, EMB);
    attn_prep_kernel<<<ROWS, 256, 0, stream>>>(qkvB, Qb, Kb, Vtb);
    attention_kernel<<<dim3(SEQ / 16, NH, BSZ), 32, 0, stream>>>(Qb, Kb, Vtb, aB);
    gemm_wmma_kernel<1><<<dim3(EMB / 128, ROWS / 128), 256, 0, stream>>>(
        aB, woT + (size_t)l * EMB * EMB, bo + (size_t)l * EMB,
        hbuf, nullptr, ROWS, EMB, EMB);
    ln_bf16_kernel<<<ROWS, 256, 0, stream>>>(hbuf, ln2_g + (size_t)l * EMB,
                                             ln2_b + (size_t)l * EMB, aB);
    gemm_wmma_kernel<2><<<dim3(FF4 / 128, ROWS / 128), 256, 0, stream>>>(
        aB, w1T + (size_t)l * FF4 * EMB, b1 + (size_t)l * FF4,
        nullptr, ffB, ROWS, FF4, EMB);
    gemm_wmma_kernel<1><<<dim3(EMB / 128, ROWS / 128), 256, 0, stream>>>(
        ffB, w2T + (size_t)l * EMB * FF4, b2 + (size_t)l * EMB,
        hbuf, nullptr, ROWS, EMB, FF4);
  }

  ln_last_kernel<<<BSZ, 256, 0, stream>>>(hbuf, lnf_g, lnf_b, lastB);
  logits_kernel<<<(VSZ + 255) / 256, 256, 0, stream>>>(lastB, tok, (float*)d_out);
}
